// OneDirectionalTreeLSTM_35862976921785
// MI455X (gfx1250) — compile-verified, matrix operand order checked by reading, not run on hardware
//
#include <hip/hip_runtime.h>
#include <hip/hip_bf16.h>

// ---------------------------------------------------------------------------
// OneDirectionalTreeLSTM for MI455X (gfx1250)
//   Phase A: fused bf16 WMMA GEMM  G[8192x3072] = X[8192x1024] * [Wpx;Wx]^T + b
//   Phase B: persistent 16-WGP recurrence, Wh slice bf16-packed in LDS
//            (bank-conflict-free b64 layout), packed bf16 dot2 mat-vec,
//            per-step cross-WGP sync via agent-scope counters.
// ---------------------------------------------------------------------------

#define N_NODES 8192
#define FEAT    1024
#define HDIM    512
#define GDIM    3072            // 512 (px) + 2560 (5 gates)
#define NWG_REC 16
#define ROWS_PER_WG 160         // 5 gates * 32 h-elements
#define HALFROWS    320         // 2 threads per gate row
#define K2_HALF     128         // packed bf16 pairs per half row (256 elems)
#define HR_STRIDE   130         // 128 + 2 pad words: lane stride 2 mod 64 banks

typedef __bf16 bf16_t;
typedef bf16_t v16bf __attribute__((ext_vector_type(16)));
typedef bf16_t v8bf  __attribute__((ext_vector_type(8)));
typedef bf16_t v2bf  __attribute__((ext_vector_type(2)));
typedef float  v8f   __attribute__((ext_vector_type(8)));

__device__ __forceinline__ unsigned int f32_to_bf16_rne(float f) {
    unsigned int u = __float_as_uint(f);
    u = (u + 0x7FFFu + ((u >> 16) & 1u)) >> 16;
    return u;
}

__device__ __forceinline__ unsigned int pack_bf16_pair(float a, float b) {
    return (f32_to_bf16_rne(b) << 16) | f32_to_bf16_rne(a);
}

// packed bf16x2 dot-accumulate: acc += w.lo*h.lo + w.hi*h.hi
__device__ __forceinline__ float dot2_bf16(unsigned int w, unsigned int h, float acc) {
#if __has_builtin(__builtin_amdgcn_fdot2_f32_bf16)
    return __builtin_amdgcn_fdot2_f32_bf16(__builtin_bit_cast(v2bf, w),
                                           __builtin_bit_cast(v2bf, h),
                                           acc, false);
#else
    const float wl = __uint_as_float(w << 16);
    const float wh = __uint_as_float(w & 0xFFFF0000u);
    const float hl = __uint_as_float(h << 16);
    const float hh = __uint_as_float(h & 0xFFFF0000u);
    acc = fmaf(wl, hl, acc);
    return fmaf(wh, hh, acc);
#endif
}

// ------------------------- init: zero step counters -------------------------
__global__ void zero_counters_kernel(int* __restrict__ c, int n) {
    int i = blockIdx.x * blockDim.x + threadIdx.x;
    if (i < n) c[i] = 0;
}

// ------------------------- fp32 -> bf16 conversion --------------------------
__global__ void cvt_f32_bf16_kernel(const float* __restrict__ src,
                                    unsigned short* __restrict__ dst, int n) {
    int i = blockIdx.x * blockDim.x + threadIdx.x;
    if (i < n) dst[i] = (unsigned short)f32_to_bf16_rne(src[i]);
}

// ------------------------- WMMA GEMM: G = X * Wcomb^T + bias ----------------
// block = 256 threads (8 waves, 2x4 wave grid) -> 32x64 output tile
// grid  = (3072/64, 8192/32) = (48, 256)
__global__ __launch_bounds__(256)
void gemm_bf16_wmma_kernel(const unsigned short* __restrict__ Abf,   // [8192][1024]
                           const unsigned short* __restrict__ Bbf,   // [3072][1024]
                           const float* __restrict__ bpx,            // [512]
                           const float* __restrict__ bx,             // [2560]
                           float* __restrict__ G) {                  // [8192][3072]
    const int lane = threadIdx.x & 31;
    const int wave = threadIdx.x >> 5;
    const int wm = wave >> 2, wn = wave & 3;
    const int mTile = blockIdx.y * 32 + wm * 16;
    const int nTile = blockIdx.x * 64 + wn * 16;
    const int mr = lane & 15;       // row of A / col of B owned by this lane
    const int hi = lane >> 4;       // K-half selector per ISA A/B layout

    const unsigned short* Ap = Abf + (size_t)(mTile + mr) * FEAT;
    const unsigned short* Bp = Bbf + (size_t)(nTile + mr) * FEAT;

    v8f acc = {};
#pragma unroll 4
    for (int kb = 0; kb < FEAT / 32; ++kb) {
        const int koff = kb * 32 + hi * 8;
        __builtin_prefetch(Ap + koff + 256, 0, 1);   // -> global_prefetch_b8
        __builtin_prefetch(Bp + koff + 256, 0, 1);
        v8bf alo = *(const v8bf*)(Ap + koff);
        v8bf ahi = *(const v8bf*)(Ap + koff + 16);
        v8bf blo = *(const v8bf*)(Bp + koff);
        v8bf bhi = *(const v8bf*)(Bp + koff + 16);
        v16bf a = __builtin_shufflevector(alo, ahi, 0, 1, 2, 3, 4, 5, 6, 7,
                                          8, 9, 10, 11, 12, 13, 14, 15);
        v16bf b = __builtin_shufflevector(blo, bhi, 0, 1, 2, 3, 4, 5, 6, 7,
                                          8, 9, 10, 11, 12, 13, 14, 15);
        acc = __builtin_amdgcn_wmma_f32_16x16x32_bf16(
            /*neg_a=*/false, a, /*neg_b=*/false, b,
            /*c_mod=*/(short)0, acc, /*reuse_a=*/false, /*reuse_b=*/false);
    }

    const int col = nTile + mr;
    const float bias = (col < HDIM) ? bpx[col] : bx[col - HDIM];
    const int rowBase = mTile + hi * 8;
#pragma unroll
    for (int v = 0; v < 8; ++v)
        G[(size_t)(rowBase + v) * GDIM + col] = acc[v] + bias;
}

// ------------------------- persistent recurrence ----------------------------
// 16 workgroups x 320 threads (10 waves). WG wg owns h-slice [wg*32, wg*32+32)
// and its 160 gate rows {g*512 + wg*32 + r}. Each gate row is split across
// 2 threads (256-element half-dots). Wh slice lives bf16-packed in LDS with a
// 130-word half-row stride (conflict-free b64 loads). h is packed to bf16
// pairs once per step and read as broadcast b64s.
__device__ __forceinline__ float sigmoidf_(float x) {
    return 1.0f / (1.0f + __expf(-x));
}

__global__ __launch_bounds__(HALFROWS)
void tree_lstm_rec_kernel(const int* __restrict__ parents,
                          const float* __restrict__ Wh,   // [2560][512] fp32
                          const float* __restrict__ bh,   // [2560]
                          const float* __restrict__ G,    // [8192][3072]
                          float* __restrict__ Hb,         // d_out [8192][512]
                          float* __restrict__ c_all,      // ws    [8192][512]
                          int* __restrict__ counter,      // ws    [8192]
                          int nwg, int nsteps) {
    extern __shared__ unsigned char smem[];
    unsigned int* Wlds  = (unsigned int*)smem;                    // 320*130 words
    unsigned int* hbl   = Wlds + HALFROWS * HR_STRIDE;            // 256 words
    float*        gpart = (float*)(hbl + 256);                    // 320 floats
    float*        gls   = gpart + HALFROWS;                       // 160 floats

    const int tid  = threadIdx.x;
    const int wg   = blockIdx.x;
    const int row  = tid >> 1;                 // gate row 0..159
    const int half = tid & 1;                  // which 256-elem half of the dot
    const int g    = row >> 5, r = row & 31;
    const int grow = g * HDIM + wg * 32 + r;   // global gate row in [0,2560)

    // ---- one-time: load this WG's Wh slice, pack to bf16 pairs in LDS ----
    float bhv = 0.0f;
    if (half == 0) bhv = bh[grow];
    {
        const float* wr = Wh + (size_t)grow * HDIM + half * 256;
        for (int k2 = 0; k2 < K2_HALF; ++k2)
            Wlds[tid * HR_STRIDE + k2] = pack_bf16_pair(wr[2 * k2], wr[2 * k2 + 1]);
    }
    __syncthreads();

    for (int t = 0; t < nsteps; ++t) {
        const int p = parents[t];

        // ---- wait until all WGs have published step p ----
        if (tid == 0 && p >= 0) {
            while (__hip_atomic_load(&counter[p], __ATOMIC_ACQUIRE,
                                     __HIP_MEMORY_SCOPE_AGENT) < nwg)
                __builtin_amdgcn_s_sleep(1);
        }
        __syncthreads();

        // ---- broadcast parent h into LDS as packed bf16 pairs ----
        if (tid < 256) {
            unsigned int w = 0u;
            if (p >= 0) {
                const float* hp = Hb + (size_t)p * HDIM + tid * 2;
                const float h0 = __hip_atomic_load(&hp[0], __ATOMIC_RELAXED,
                                                   __HIP_MEMORY_SCOPE_AGENT);
                const float h1 = __hip_atomic_load(&hp[1], __ATOMIC_RELAXED,
                                                   __HIP_MEMORY_SCOPE_AGENT);
                w = pack_bf16_pair(h0, h1);
            }
            hbl[tid] = w;
        }
        __syncthreads();

        // ---- mat-vec: 320 half-dots of length 256 from LDS ----
        {
            float acc = (half == 0) ? bhv + G[(size_t)t * GDIM + HDIM + grow] : 0.0f;
            const unsigned int* wp = Wlds + tid * HR_STRIDE;
            const unsigned int* hp = hbl + half * K2_HALF;
#pragma unroll 4
            for (int k4 = 0; k4 < K2_HALF / 2; ++k4) {
                const uint2 w2 = *(const uint2*)(wp + 2 * k4);   // ds_load_b64
                const uint2 h2 = *(const uint2*)(hp + 2 * k4);   // broadcast b64
                acc = dot2_bf16(w2.x, h2.x, acc);
                acc = dot2_bf16(w2.y, h2.y, acc);
            }
            gpart[tid] = acc;
        }
        __syncthreads();
        if (tid < ROWS_PER_WG)
            gls[tid] = gpart[2 * tid] + gpart[2 * tid + 1];
        __syncthreads();

        // ---- gate nonlinearities + state update for owned 32 h-elements ---
        if (tid < 32) {
            const int slice = wg * 32 + tid;
            const float gi = gls[tid];
            const float go = gls[32 + tid];
            const float gf = gls[64 + tid];
            const float gu = gls[96 + tid];
            const float gr = gls[128 + tid];
            const float cprev = (p >= 0)
                ? __hip_atomic_load(&c_all[(size_t)p * HDIM + slice],
                                    __ATOMIC_RELAXED, __HIP_MEMORY_SCOPE_AGENT)
                : 0.0f;
            const float i_ = sigmoidf_(gi);
            const float o_ = sigmoidf_(go);
            const float f_ = sigmoidf_(gf);
            const float r_ = sigmoidf_(gr);
            const float u_ = tanhf(gu);
            const float c  = i_ * u_ + f_ * cprev;
            const float hh = o_ * tanhf(c);
            const float px = G[(size_t)t * GDIM + slice];
            const float h  = r_ * hh + (1.0f - r_) * px;
            __hip_atomic_store(&c_all[(size_t)t * HDIM + slice], c,
                               __ATOMIC_RELAXED, __HIP_MEMORY_SCOPE_AGENT);
            __hip_atomic_store(&Hb[(size_t)t * HDIM + slice], h,
                               __ATOMIC_RELAXED, __HIP_MEMORY_SCOPE_AGENT);
        }
        __threadfence();        // make slice stores visible at device scope
        __syncthreads();
        if (tid == 0)
            __hip_atomic_fetch_add(&counter[t], 1, __ATOMIC_RELEASE,
                                   __HIP_MEMORY_SCOPE_AGENT);
    }
}

// ---------------------------------------------------------------------------
extern "C" void kernel_launch(void* const* d_in, const int* in_sizes, int n_in,
                              void* d_out, int out_size, void* d_ws, size_t ws_size,
                              hipStream_t stream) {
    (void)in_sizes; (void)n_in; (void)out_size; (void)ws_size;

    const float* features = (const float*)d_in[0];   // [8192*1024]
    const int*   parents  = (const int*)d_in[1];     // [8192]
    const float* Wpx      = (const float*)d_in[2];   // [512*1024]
    const float* bpx      = (const float*)d_in[3];   // [512]
    const float* Wx       = (const float*)d_in[4];   // [2560*1024]
    const float* bx       = (const float*)d_in[5];   // [2560]
    const float* Wh       = (const float*)d_in[6];   // [2560*512]
    const float* bh       = (const float*)d_in[7];   // [2560]
    float* Hb = (float*)d_out;                       // [8192*512]

    // ---- workspace carve-up ----
    char* base = (char*)d_ws;
    size_t off = 0;
    unsigned short* featB  = (unsigned short*)(base + off); off += (size_t)N_NODES * FEAT * 2;  // 16 MB
    unsigned short* WcombB = (unsigned short*)(base + off); off += (size_t)GDIM * FEAT * 2;     //  6 MB
    float* G      = (float*)(base + off); off += (size_t)N_NODES * GDIM * 4;                    // 96 MB
    float* c_all  = (float*)(base + off); off += (size_t)N_NODES * HDIM * 4;                    // 16 MB
    int*   ctr    = (int*)(base + off);   off += (size_t)N_NODES * 4;                           // 32 KB

    // 1) reset per-step counters (replay safety)
    zero_counters_kernel<<<(N_NODES + 255) / 256, 256, 0, stream>>>(ctr, N_NODES);

    // 2) bf16 conversions: features, and combined weight [Wpx; Wx]
    {
        int n = N_NODES * FEAT;
        cvt_f32_bf16_kernel<<<(n + 255) / 256, 256, 0, stream>>>(features, featB, n);
        n = HDIM * FEAT;
        cvt_f32_bf16_kernel<<<(n + 255) / 256, 256, 0, stream>>>(Wpx, WcombB, n);
        n = 5 * HDIM * FEAT;
        cvt_f32_bf16_kernel<<<(n + 255) / 256, 256, 0, stream>>>(
            Wx, WcombB + (size_t)HDIM * FEAT, n);
    }

    // 3) fused WMMA GEMM: G = X * Wcomb^T + [bpx; bx]
    gemm_bf16_wmma_kernel<<<dim3(GDIM / 64, N_NODES / 32), 256, 0, stream>>>(
        featB, WcombB, bpx, bx, G);

    // 4) persistent chain recurrence across 16 WGPs
    const size_t lds_bytes = (size_t)HALFROWS * HR_STRIDE * 4   // packed Wh slice
                           + 256 * 4                            // packed h
                           + HALFROWS * 4                       // half-dot partials
                           + ROWS_PER_WG * 4;                   // gate scratch
    (void)hipFuncSetAttribute((const void*)tree_lstm_rec_kernel,
                              hipFuncAttributeMaxDynamicSharedMemorySize,
                              (int)lds_bytes);
    tree_lstm_rec_kernel<<<NWG_REC, HALFROWS, lds_bytes, stream>>>(
        parents, Wh, bh, G, Hb, c_all, ctr, NWG_REC, N_NODES);
}
